// PerformerAttention_36593121362541
// MI455X (gfx1250) — matrix operand order repaired and checked by reading, not compiled, not run on hardware
//
#include <hip/hip_runtime.h>
#include <hip/hip_fp16.h>

// ---------------------------------------------------------------------------
// Performer (FAVOR+) attention for MI455X / gfx1250, wave32 WMMA f16 pipeline.
// All GEMMs are "NT": A is [M,K] row-major, B is [N,K] row-major, so both
// WMMA fragments are loaded straight from global memory with 2x b128 per lane.
// Transposed operands for later stages are produced by packed transposed
// epilogue stores. The small-output context GEMM (K=8192) is split-K x16 into
// an f32 partial buffer + fixed-order reduction for occupancy + determinism.
// ---------------------------------------------------------------------------

#define HEADS 8
#define DH 64
#define NF 256
#define DIM 512
#define NTOK 8192              // n per batch
#define BN (2 * NTOK)          // 16384 total rows
#define E3 (3 * HEADS * DH)    // 1536
#define KSPLIT 16
#define KCHUNK (NTOK / KSPLIT) // 512

typedef __attribute__((ext_vector_type(16))) _Float16 v16h;
typedef __attribute__((ext_vector_type(8)))  _Float16 v8h;
typedef __attribute__((ext_vector_type(8)))  float    v8f;
typedef __attribute__((ext_vector_type(4)))  float    v4f;

union HFrag { v16h v16; v8h h8[2]; };
union FSplit { v8f v8; v4f v4[2]; };

// A-fragment (16x32 f16, M x K): lane m = lane&15; halves j=0..15 <-> K=16*hi+j
// (same k order as frag_b).
__device__ __forceinline__ v16h frag_a(const _Float16* __restrict__ tile,
                                       size_t lda, int lane) {
    const int m = lane & 15, hi = lane >> 4;
    const _Float16* row = tile + (size_t)m * lda;
    HFrag f;
    f.h8[0] = *(const v8h*)(row + 16 * hi);
    f.h8[1] = *(const v8h*)(row + 16 * hi + 8);
    return f.v16;
}

// B-fragment (32x16 f16, K x N) from a [N,K] row-major tile: lane n = lane&15;
// halves j=0..15 <-> K = 16*hi + j. (analog of sparse 64x16 B layout)
__device__ __forceinline__ v16h frag_b(const _Float16* __restrict__ tile,
                                       size_t ldb, int lane) {
    const int n = lane & 15, hi = lane >> 4;
    const _Float16* row = tile + (size_t)n * ldb + 16 * hi;
    HFrag f;
    f.h8[0] = *(const v8h*)(row);
    f.h8[1] = *(const v8h*)(row + 8);
    return f.v16;
}

__device__ __forceinline__ v8f wmma_kloop(const _Float16* __restrict__ A, size_t lda,
                                          const _Float16* __restrict__ B, size_t ldb,
                                          int K, int lane) {
    v8f acc = {};
    for (int k = 0; k < K; k += 32) {
        v16h a = frag_a(A + k, lda, lane);
        v16h b = frag_b(B + k, ldb, lane);
        acc = __builtin_amdgcn_wmma_f32_16x16x32_f16(
            /*neg_a=*/false, a, /*neg_b=*/false, b,
            /*c_mod=*/(short)0, acc, /*reuse_a=*/false, /*reuse_b=*/false);
    }
    return acc;
}

// ---------------------------------------------------------------------------

__global__ void cvt_f32_f16(const float* __restrict__ in, _Float16* __restrict__ out,
                            int nElem) {
    int i = blockIdx.x * blockDim.x + threadIdx.x;
    if (i < nElem) out[i] = (_Float16)in[i];
}

// qkv[16384,1536] = xh[16384,512] @ Wqkvh^T   (Wqkvh is [1536,512] = [N,K])
__global__ void gemm_qkv(const _Float16* __restrict__ xh,
                         const _Float16* __restrict__ wqkvh,
                         _Float16* __restrict__ qkvh) {
    const int lane = threadIdx.x & 31, w = threadIdx.x >> 5;
    const int tm = blockIdx.x;                 // 0..1023
    const int tn = blockIdx.y * 8 + w;         // 0..95
    v8f acc = wmma_kloop(xh + (size_t)tm * 16 * DIM, DIM,
                         wqkvh + (size_t)tn * 16 * DIM, DIM, DIM, lane);
    const int nn = lane & 15, hi = lane >> 4;
#pragma unroll
    for (int r = 0; r < 8; ++r) {
        size_t m = (size_t)tm * 16 + r + 8 * hi;
        qkvh[m * E3 + tn * 16 + nn] = (_Float16)acc[r];
    }
}

// q' = exp(scale * Q_{b,h} @ proj^T)   -> qprime[bh][n][f] (row-major)
__global__ void gemm_featq(const _Float16* __restrict__ qkvh,
                           const _Float16* __restrict__ projh,
                           _Float16* __restrict__ qprime) {
    const int lane = threadIdx.x & 31, w = threadIdx.x >> 5;
    const int tm = blockIdx.x;                 // 0..511 (n tiles)
    const int tn = blockIdx.y * 8 + w;         // 0..15  (f tiles)
    const int bh = blockIdx.z, b = bh >> 3, h = bh & 7;
    const _Float16* A = qkvh + ((size_t)(b * NTOK + tm * 16)) * E3 + h * DH;
    const _Float16* B = projh + (size_t)tn * 16 * DH;
    v8f acc = wmma_kloop(A, E3, B, DH, DH, lane);
    const int nn = lane & 15, hi = lane >> 4;
    _Float16* outb = qprime + (size_t)bh * NTOK * NF;
#pragma unroll
    for (int r = 0; r < 8; ++r) {
        int m = tm * 16 + r + 8 * hi;
        outb[(size_t)m * NF + tn * 16 + nn] = (_Float16)__expf(0.125f * acc[r]);
    }
}

// k' = exp(K_{b,h} @ proj^T) stored TRANSPOSED: kprimeT[bh][f][n]
__global__ void gemm_featk(const _Float16* __restrict__ qkvh,
                           const _Float16* __restrict__ projh,
                           _Float16* __restrict__ kprimeT) {
    const int lane = threadIdx.x & 31, w = threadIdx.x >> 5;
    const int tm = blockIdx.x;                 // n tiles
    const int tn = blockIdx.y * 8 + w;         // f tiles
    const int bh = blockIdx.z, b = bh >> 3, h = bh & 7;
    const _Float16* A = qkvh + ((size_t)(b * NTOK + tm * 16)) * E3 + DIM + h * DH;
    const _Float16* B = projh + (size_t)tn * 16 * DH;
    v8f acc = wmma_kloop(A, E3, B, DH, DH, lane);
    const int nn = lane & 15, hi = lane >> 4;
    v8h pk;
#pragma unroll
    for (int r = 0; r < 8; ++r) pk[r] = (_Float16)__expf(acc[r]);
    // element (n = tm*16 + r + 8*hi, f = tn*16 + nn); 8 contiguous n per lane
    _Float16* dst = kprimeT + (size_t)bh * NF * NTOK
                  + (size_t)(tn * 16 + nn) * NTOK + tm * 16 + 8 * hi;
    *(v8h*)dst = pk;
}

// vT[bh][d][n] <- qkv v-part
__global__ void make_vT(const _Float16* __restrict__ qkvh, _Float16* __restrict__ vT) {
    int idx = blockIdx.x * 256 + threadIdx.x;   // 16*64*8192
    int n = idx & (NTOK - 1);
    int t = idx >> 13;
    int d = t & (DH - 1);
    int bh = t >> 6, b = bh >> 3, h = bh & 7;
    vT[idx] = qkvh[((size_t)(b * NTOK + n)) * E3 + 2 * DIM + h * DH + d];
}

// ksum[bh*256+f] = sum_n kprimeT row
__global__ void ksum_kernel(const _Float16* __restrict__ kprimeT, float* __restrict__ ksum) {
    const int row = blockIdx.x;                 // 0..4095
    const _Float16* p = kprimeT + (size_t)row * NTOK;
    float s = 0.f;
    for (int i = threadIdx.x; i < NTOK; i += 256) s += (float)p[i];
    __shared__ float red[256];
    red[threadIdx.x] = s;
    __syncthreads();
    for (int off = 128; off > 0; off >>= 1) {
        if (threadIdx.x < off) red[threadIdx.x] += red[threadIdx.x + off];
        __syncthreads();
    }
    if (threadIdx.x == 0) ksum[row] = red[0];
}

// dinv[bh][n] = 1/(q'[n,:] . ksum[bh,:] + 1e-6)
__global__ void dinv_kernel(const _Float16* __restrict__ qprime,
                            const float* __restrict__ ksum, float* __restrict__ dinv) {
    int idx = blockIdx.x * 256 + threadIdx.x;   // 16*8192
    int n = idx & (NTOK - 1);
    int bh = idx >> 13;
    const _Float16* qp = qprime + ((size_t)bh * NTOK + n) * NF;
    const float* ks = ksum + bh * NF;
    float acc = 0.f;
    for (int f = 0; f < NF; ++f) acc += (float)qp[f] * ks[f];
    dinv[idx] = 1.0f / (acc + 1e-6f);
}

// context split-K partials: part[split][bh][dh][f] f32
// per (b,h): ctx[f=256, dh=64] = k'^T @ v over K chunk of 512
__global__ void gemm_ctx_split(const _Float16* __restrict__ kprimeT,
                               const _Float16* __restrict__ vT,
                               float* __restrict__ ctx_part) {
    const int lane = threadIdx.x & 31, w = threadIdx.x >> 5;
    const int bh = blockIdx.z;
    const int split = blockIdx.y;              // 0..15
    const int tm = blockIdx.x * 2 + (w >> 2);  // f tiles 0..15
    const int tn = w & 3;                      // dh tiles 0..3
    const _Float16* A = kprimeT + (size_t)bh * NF * NTOK + (size_t)tm * 16 * NTOK
                      + (size_t)split * KCHUNK;
    const _Float16* B = vT + (size_t)bh * DH * NTOK + (size_t)tn * 16 * NTOK
                      + (size_t)split * KCHUNK;
    FSplit acc;
    acc.v8 = wmma_kloop(A, NTOK, B, NTOK, KCHUNK, lane);
    const int nn = lane & 15, hi = lane >> 4;
    // element (f = tm*16 + r + 8*hi, d = tn*16 + nn); 8 contiguous f per lane
    float* dst = ctx_part + (((size_t)split * 16 + bh) * DH + (tn * 16 + nn)) * NF
               + tm * 16 + 8 * hi;
    *(v4f*)dst       = acc.v4[0];
    *(v4f*)(dst + 4) = acc.v4[1];
}

// fixed-order sum of KSPLIT partials -> ctxT[bh][d][f] f16 (deterministic)
__global__ void ctx_reduce(const float* __restrict__ ctx_part,
                           _Float16* __restrict__ ctxT) {
    int idx = blockIdx.x * 256 + threadIdx.x;   // 16*64*256 = 262144
    float s = 0.f;
#pragma unroll
    for (int sp = 0; sp < KSPLIT; ++sp)
        s += ctx_part[(size_t)sp * 16 * DH * NF + idx];
    ctxT[idx] = (_Float16)s;
}

// out = (q' @ context) * Dinv, head-concat -> outh[16384][512] f16
__global__ void gemm_out(const _Float16* __restrict__ qprime,
                         const _Float16* __restrict__ ctxT,
                         const float* __restrict__ dinv,
                         _Float16* __restrict__ outh) {
    const int lane = threadIdx.x & 31, w = threadIdx.x >> 5;
    const int bh = blockIdx.z, b = bh >> 3, h = bh & 7;
    const int tm = blockIdx.x * 2 + (w >> 2);  // n tiles 0..511 over grid.x=256
    const int tn = w & 3;                      // dh tiles
    const _Float16* A = qprime + (size_t)bh * NTOK * NF + (size_t)tm * 16 * NF;
    const _Float16* B = ctxT + (size_t)bh * DH * NF + (size_t)tn * 16 * NF;
    v8f acc = wmma_kloop(A, NF, B, NF, NF, lane);
    const int nn = lane & 15, hi = lane >> 4;
    const float* dv = dinv + bh * NTOK;
#pragma unroll
    for (int r = 0; r < 8; ++r) {
        int m = tm * 16 + r + 8 * hi;
        float val = acc[r] * dv[m];
        outh[((size_t)(b * NTOK + m)) * DIM + h * DH + tn * 16 + nn] = (_Float16)val;
    }
}

// final[16384,512] f32 = outh @ Wout^T + bout
__global__ void gemm_final(const _Float16* __restrict__ outh,
                           const _Float16* __restrict__ wouth,
                           const float* __restrict__ bout,
                           float* __restrict__ out) {
    const int lane = threadIdx.x & 31, w = threadIdx.x >> 5;
    const int tm = blockIdx.x;                 // 0..1023
    const int tn = blockIdx.y * 8 + w;         // 0..31
    v8f acc = wmma_kloop(outh + (size_t)tm * 16 * DIM, DIM,
                         wouth + (size_t)tn * 16 * DIM, DIM, DIM, lane);
    const int nn = lane & 15, hi = lane >> 4;
    const int col = tn * 16 + nn;
    const float bb = bout[col];
#pragma unroll
    for (int r = 0; r < 8; ++r) {
        size_t m = (size_t)tm * 16 + r + 8 * hi;
        out[m * DIM + col] = acc[r] + bb;
    }
}

// ---------------------------------------------------------------------------

extern "C" void kernel_launch(void* const* d_in, const int* in_sizes, int n_in,
                              void* d_out, int out_size, void* d_ws, size_t ws_size,
                              hipStream_t stream) {
    const float* x    = (const float*)d_in[0];   // [2,8192,512]
    const float* Wqkv = (const float*)d_in[1];   // [1536,512]
    const float* Wout = (const float*)d_in[2];   // [512,512]
    const float* bout = (const float*)d_in[3];   // [512]
    const float* proj = (const float*)d_in[4];   // [256,64]
    float* outp = (float*)d_out;

    // workspace carving (256B aligned slots)
    char* ws = (char*)d_ws;
    size_t off = 0;
    auto carve = [&](size_t bytes) -> char* {
        char* p = ws + off;
        off += (bytes + 255) & ~(size_t)255;
        return p;
    };
    _Float16* xh       = (_Float16*)carve((size_t)BN * DIM * 2);       // 16 MB
    _Float16* wqkvh    = (_Float16*)carve((size_t)E3 * DIM * 2);       // 1.5 MB
    _Float16* wouth    = (_Float16*)carve((size_t)DIM * DIM * 2);      // 0.5 MB
    _Float16* projh    = (_Float16*)carve((size_t)NF * DH * 2);        // 32 KB
    _Float16* qkvh     = (_Float16*)carve((size_t)BN * E3 * 2);        // 48 MB
    _Float16* qprime   = (_Float16*)carve((size_t)16 * NTOK * NF * 2); // 64 MB
    _Float16* kprimeT  = (_Float16*)carve((size_t)16 * NF * NTOK * 2); // 64 MB
    _Float16* vT       = (_Float16*)carve((size_t)16 * DH * NTOK * 2); // 16 MB
    _Float16* ctxT     = (_Float16*)carve((size_t)16 * DH * NF * 2);   // 0.5 MB
    float*    ctx_part = (float*)carve((size_t)KSPLIT * 16 * DH * NF * 4); // 16 MB
    _Float16* outh     = (_Float16*)carve((size_t)BN * DIM * 2);       // 16 MB
    float*    ksum     = (float*)carve((size_t)16 * NF * 4);           // 16 KB
    float*    dinv     = (float*)carve((size_t)16 * NTOK * 4);         // 512 KB

    const dim3 blk(256);

    // f32 -> f16 conversions
    cvt_f32_f16<<<(BN * DIM + 255) / 256, blk, 0, stream>>>(x, xh, BN * DIM);
    cvt_f32_f16<<<(E3 * DIM + 255) / 256, blk, 0, stream>>>(Wqkv, wqkvh, E3 * DIM);
    cvt_f32_f16<<<(DIM * DIM + 255) / 256, blk, 0, stream>>>(Wout, wouth, DIM * DIM);
    cvt_f32_f16<<<(NF * DH + 255) / 256, blk, 0, stream>>>(proj, projh, NF * DH);

    // qkv projection
    gemm_qkv<<<dim3(BN / 16, E3 / 128, 1), blk, 0, stream>>>(xh, wqkvh, qkvh);

    // v transpose + feature maps
    make_vT<<<(16 * DH * NTOK) / 256, blk, 0, stream>>>(qkvh, vT);
    gemm_featq<<<dim3(NTOK / 16, NF / 128, 16), blk, 0, stream>>>(qkvh, projh, qprime);
    gemm_featk<<<dim3(NTOK / 16, NF / 128, 16), blk, 0, stream>>>(qkvh, projh, kprimeT);

    // normalizer
    ksum_kernel<<<16 * NF, blk, 0, stream>>>(kprimeT, ksum);
    dinv_kernel<<<(16 * NTOK) / 256, blk, 0, stream>>>(qprime, ksum, dinv);

    // context (split-K for occupancy) + deterministic reduce
    gemm_ctx_split<<<dim3(NF / 32, KSPLIT, 16), blk, 0, stream>>>(kprimeT, vT, ctx_part);
    ctx_reduce<<<(16 * DH * NF) / 256, blk, 0, stream>>>(ctx_part, ctxT);

    // output GEMM with Dinv epilogue
    gemm_out<<<dim3(NTOK / 32, 1, 16), blk, 0, stream>>>(qprime, ctxT, dinv, outh);

    // output projection + bias
    gemm_final<<<dim3(BN / 16, DIM / 128, 1), blk, 0, stream>>>(outh, wouth, bout, outp);

    (void)in_sizes; (void)n_in; (void)out_size; (void)ws_size;
}
